// _PINNSpaceTimeMPLayer_39994735460753
// MI455X (gfx1250) — matrix operand order repaired and checked by reading, not compiled, not run on hardware
//
#include <hip/hip_runtime.h>
#include <hip/hip_bf16.h>
#include <math.h>

typedef _Float16 v16h __attribute__((ext_vector_type(16)));
typedef _Float16 v8h  __attribute__((ext_vector_type(8)));
typedef _Float16 v4h  __attribute__((ext_vector_type(4)));
typedef float    v8f  __attribute__((ext_vector_type(8)));
typedef float    v4f  __attribute__((ext_vector_type(4)));

#define NB   4
#define NT   128
#define NXD  512
#define DD   64
#define KXR  3
#define DELTA_C 0.01f
#define THR_C   0.1f
#define F16MAX  65504.f

// f16 weight workspace offsets (in halves), row-major [out][Kpad]
#define OFF_SP1 0u        // 128 x 160 (133 padded)
#define OFF_SP2 20480u    // 128 x 128
#define OFF_SP3 36864u    //  64 x 128
#define OFF_TP1 45056u    // 128 x 160 (132 padded)
#define OFF_TP2 65536u    // 128 x 128
#define OFF_TP3 81920u    //  64 x 128
#define OFF_UP1 90112u    // 128 x 128
#define OFF_UP2 106496u   // 128 x 128
#define OFF_UP3 122880u   //  64 x 128
#define OFF_WL  131072u   //  64 x 64
#define OFF_H16 135168u   // optional f16 copy of h (B*NT*NX*D halves)

// LDS slab strides (halves), padded so the 16 A-fragment rows hit disjoint
// bank groups (row stride in dwords mod 64 generates distinct 4-dword spans):
#define LD_IN 168   // 336B/row -> 84 dw, 84%64=20 : conflict-free
#define LD_H1 136   // 272B/row -> 68 dw, 68%64=4  : conflict-free
#define LD_H2 136
#define LD_HC 72    // 144B/row -> 36 dw, 36%64=36 : conflict-free

__device__ __forceinline__ int clampi(int v, int lo, int hi) {
  return v < lo ? lo : (v > hi ? hi : v);
}

__device__ __forceinline__ float gelu_exact(float x) {
  return 0.5f * x * (1.0f + erff(x * 0.70710678118654752440f));
}

// A fragment (16x32 f16) from row-major f16 LDS, row stride lds halves.
// contiguous k order, same as ld_bfrag: lane holds row m=lane&15; element e -> k=(lane/16)*16+e
__device__ __forceinline__ v16h ld_afrag(const _Float16* base, int lds, int lane) {
  const int r  = lane & 15;
  const int h8 = (lane >> 4) << 4;
  const _Float16* p = base + r * lds + h8;
  v8h lo = *(const v8h*)(p);
  v8h hi = *(const v8h*)(p + 8);
  v16h f;
#pragma unroll
  for (int e = 0; e < 8; ++e) { f[e] = lo[e]; f[8 + e] = hi[e]; }
  return f;
}

// B fragment (32x16 f16) from row-major [N][Kpad] f16 weights (global/L2):
// lane holds column n=lane&15; element e -> k=(lane/16)*16+e (one 32B load)
__device__ __forceinline__ v16h ld_bfrag(const _Float16* Wn0, int Kpad, int kbase, int lane) {
  const int n  = lane & 15;
  const int ks = (lane >> 4) << 4;
  return *(const v16h*)(Wn0 + (size_t)n * Kpad + kbase + ks);
}

// GEMM layer: dst[16][N] f16 (row stride ldd) = gelu(src x W^T + bias)
template <int KC, int N>
__device__ __forceinline__ void layer_gelu(const _Float16* src, int lds,
                                           const _Float16* W16, int Kpad,
                                           const float* bias, _Float16* dst, int ldd,
                                           int lane) {
  v16h a[KC];
#pragma unroll
  for (int kc = 0; kc < KC; ++kc) a[kc] = ld_afrag(src + kc * 32, lds, lane);
  const int n0 = lane & 15, mhi = (lane >> 4) << 3;
#pragma unroll
  for (int nt = 0; nt < N / 16; ++nt) {
    v8f acc = {0.f, 0.f, 0.f, 0.f, 0.f, 0.f, 0.f, 0.f};
#pragma unroll
    for (int kc = 0; kc < KC; ++kc) {
      v16h bf = ld_bfrag(W16 + (size_t)(nt * 16) * Kpad, Kpad, kc * 32, lane);
      acc = __builtin_amdgcn_wmma_f32_16x16x32_f16(false, a[kc], false, bf,
                                                   (short)0, acc, false, false);
    }
    const int n = nt * 16 + n0;
    const float bv = bias[n];
#pragma unroll
    for (int v = 0; v < 8; ++v) {
      float xv = acc[v] + bv;
      dst[(v + mhi) * ldd + n] = (_Float16)gelu_exact(xv);
    }
  }
}

// message layer: register agg[4] (C/D fragment layout) += gate(src x W^T + bias)
template <int KC>
__device__ __forceinline__ void layer_msg(const _Float16* src, int lds,
                                          const _Float16* W16, int Kpad,
                                          const float* bias, v8f (&agg)[4],
                                          int shkm, int lane) {
  v16h a[KC];
#pragma unroll
  for (int kc = 0; kc < KC; ++kc) a[kc] = ld_afrag(src + kc * 32, lds, lane);
  const int n0 = lane & 15;
#pragma unroll
  for (int nt = 0; nt < 4; ++nt) {
    v8f acc = {0.f, 0.f, 0.f, 0.f, 0.f, 0.f, 0.f, 0.f};
#pragma unroll
    for (int kc = 0; kc < KC; ++kc) {
      v16h bf = ld_bfrag(W16 + (size_t)(nt * 16) * Kpad, Kpad, kc * 32, lane);
      acc = __builtin_amdgcn_wmma_f32_16x16x32_f16(false, a[kc], false, bf,
                                                   (short)0, acc, false, false);
    }
    const float bv = bias[nt * 16 + n0];
#pragma unroll
    for (int v = 0; v < 8; ++v) {
      float val = acc[v] + bv;
      float g = ((shkm >> v) & 1) ? fminf(fmaxf(val, 0.f), DELTA_C) : val;
      agg[nt][v] += g;
    }
  }
}

// plain linear into register fragments: loc[4] = src x W^T + bias
template <int KC>
__device__ __forceinline__ void layer_lin(const _Float16* src, int lds,
                                          const _Float16* W16, int Kpad,
                                          const float* bias, v8f (&loc)[4], int lane) {
  v16h a[KC];
#pragma unroll
  for (int kc = 0; kc < KC; ++kc) a[kc] = ld_afrag(src + kc * 32, lds, lane);
  const int n0 = lane & 15;
#pragma unroll
  for (int nt = 0; nt < 4; ++nt) {
    v8f acc = {0.f, 0.f, 0.f, 0.f, 0.f, 0.f, 0.f, 0.f};
#pragma unroll
    for (int kc = 0; kc < KC; ++kc) {
      v16h bf = ld_bfrag(W16 + (size_t)(nt * 16) * Kpad, Kpad, kc * 32, lane);
      acc = __builtin_amdgcn_wmma_f32_16x16x32_f16(false, a[kc], false, bf,
                                                   (short)0, acc, false, false);
    }
    const float bv = bias[nt * 16 + n0];
#pragma unroll
    for (int v = 0; v < 8; ++v) loc[nt][v] = acc[v] + bv;
  }
}

// final layer: out = gelu(src x W^T + bias + loc), fp32 store to global
template <int KC>
__device__ __forceinline__ void layer_final(const _Float16* src, int lds,
                                            const _Float16* W16, int Kpad,
                                            const float* bias, const v8f (&loc)[4],
                                            float* out, long pbase, int lane) {
  v16h a[KC];
#pragma unroll
  for (int kc = 0; kc < KC; ++kc) a[kc] = ld_afrag(src + kc * 32, lds, lane);
  const int n0 = lane & 15, mhi = (lane >> 4) << 3;
#pragma unroll
  for (int nt = 0; nt < 4; ++nt) {
    v8f acc = {0.f, 0.f, 0.f, 0.f, 0.f, 0.f, 0.f, 0.f};
#pragma unroll
    for (int kc = 0; kc < KC; ++kc) {
      v16h bf = ld_bfrag(W16 + (size_t)(nt * 16) * Kpad, Kpad, kc * 32, lane);
      acc = __builtin_amdgcn_wmma_f32_16x16x32_f16(false, a[kc], false, bf,
                                                   (short)0, acc, false, false);
    }
    const int n = nt * 16 + n0;
    const float bv = bias[n];
#pragma unroll
    for (int v = 0; v < 8; ++v) {
      float val = acc[v] + bv + loc[nt][v];
      out[(pbase + v + mhi) * DD + n] = gelu_exact(val);
    }
  }
}

// ---------------- prep: f32 weights -> f16 with K zero-padding ----------------
__global__ void pinn_wprep_f16(const float* __restrict__ W, _Float16* __restrict__ dst,
                               int rows, int cols, int ldd) {
  int i = blockIdx.x * blockDim.x + threadIdx.x;
  if (i >= rows * ldd) return;
  int r = i / ldd, c = i - r * ldd;
  dst[i] = (c < cols) ? (_Float16)W[(size_t)r * cols + c] : (_Float16)0.f;
}

// ---------------- prep: h f32 -> f16 (vectorized) ----------------
__global__ void pinn_hprep_f16(const float* __restrict__ src, _Float16* __restrict__ dst,
                               int n4) {
  int i = blockIdx.x * blockDim.x + threadIdx.x;
  if (i >= n4) return;
  v4f f = *(const v4f*)(src + (size_t)i * 4);
  v4h o;
#pragma unroll
  for (int e = 0; e < 4; ++e) o[e] = (_Float16)f[e];
  *(v4h*)(dst + (size_t)i * 4) = o;
}

// ---------------- main fused kernel ----------------
__global__ __launch_bounds__(128) void pinn_mp_main(
    const float* __restrict__ h, const float* __restrict__ x, const float* __restrict__ t,
    const float* __restrict__ u0, const float* __restrict__ shock,
    const _Float16* __restrict__ wf, const _Float16* __restrict__ h16,
    const float* __restrict__ bsp1, const float* __restrict__ bsp2, const float* __restrict__ bsp3,
    const float* __restrict__ btp1, const float* __restrict__ btp2, const float* __restrict__ btp3,
    const float* __restrict__ bup1, const float* __restrict__ bup2, const float* __restrict__ bup3,
    const float* __restrict__ bwl,
    float* __restrict__ out) {
  __shared__ __align__(16) _Float16 sIN[4][16 * LD_IN];  // input slab / H2 ping
  __shared__ __align__(16) _Float16 sH1[4][16 * LD_H1];  // hidden-1
  __shared__ __align__(16) _Float16 sHC[4][16 * LD_HC];  // center h tile (f16)
  __shared__ float sX[64], sU0[64];

  const int tid = threadIdx.x, w = tid >> 5, lane = tid & 31;
  const int blk = blockIdx.x;
  const int b   = blk >> 10;          // NT * (NX/64) = 1024 blocks per batch
  const int rem = blk & 1023;
  const int it  = rem >> 3;
  const int xw0 = ((rem & 7) << 6) + (w << 4);   // this wave's first x index
  const long rowc  = ((long)(b * NT + it)) * NXD; // point row base at (b,it)
  const long pbase = rowc + xw0;
  const int rg0 = w << 4;

  _Float16* sINw = sIN[w];
  _Float16* sH1w = sH1[w];
  _Float16* sHCw = sHC[w];

  if (tid == 0) {  // warm L2 for the f16 weight block
    __builtin_prefetch(wf, 0, 1);
    __builtin_prefetch(wf + OFF_TP1, 0, 1);
    __builtin_prefetch(wf + OFF_UP1, 0, 1);
  }

  if (lane < 16) {
    const int gx = xw0 + lane;
    sX[rg0 + lane]  = x[b * NXD + gx];
    sU0[rg0 + lane] = u0[b * NXD + gx];
  }

  // per-lane shock mask for the 8 rows this lane's C/D fragment elements map to
  const int mhi = (lane >> 4) << 3;
  int shkm = 0;
#pragma unroll
  for (int v = 0; v < 8; ++v)
    shkm |= (shock[pbase + mhi + v] > THR_C) ? (1 << v) : 0;

  // center h tile -> LDS f16 (padded stride)
  if (h16) {
    const _Float16* hc16 = h16 + pbase * DD;
    for (int i = lane; i < 128; i += 32) {       // 16 rows x 8 chunks of 8 halves
      const int r = i >> 3, c8 = (i & 7) << 3;
      *(v8h*)&sHCw[r * LD_HC + c8] = *(const v8h*)(hc16 + r * DD + c8);
    }
  } else {
    const float* hc = h + pbase * DD;
    for (int i = lane; i < 256; i += 32) {       // 16 rows x 16 chunks of 4
      const int r = i >> 4, c4 = (i & 15) << 2;
      v4f f = *(const v4f*)(hc + r * DD + c4);
      v4h o;
#pragma unroll
      for (int e = 0; e < 4; ++e) o[e] = (_Float16)f[e];
      *(v4h*)&sHCw[r * LD_HC + c4] = o;
    }
  }
  __syncthreads();

  const float ti = t[it];
  v8f agg[4];
#pragma unroll
  for (int nt = 0; nt < 4; ++nt) agg[nt] = (v8f){0.f,0.f,0.f,0.f,0.f,0.f,0.f,0.f};

  // 7 spatial passes (j=-3..3) then 3 temporal passes (j=-2..0)
  for (int pi = 0; pi < 10; ++pi) {
    const bool spatial = (pi < 7);
    const int  j = spatial ? (pi - KXR) : (pi - 9);
    long nbase = 0;    // neighbor block base (point index) for temporal
    float tj = 0.f, reltj = 0.f;
    if (!spatial) {
      const int itj = clampi(it + j, 0, NT - 1);
      nbase = ((long)(b * NT + itj)) * NXD + xw0;
      tj = t[itj];
      reltj = tj - ti;
    }

    // cols 0..63: center h (LDS->LDS, 16B chunks)
    for (int i = lane; i < 128; i += 32) {
      const int r = i >> 3, c8 = (i & 7) << 3;
      *(v8h*)&sINw[r * LD_IN + c8] = *(const v8h*)&sHCw[r * LD_HC + c8];
    }
    // cols 64..127: neighbor h (edge-clamped)
    if (h16) {
      for (int i = lane; i < 128; i += 32) {
        const int r = i >> 3, c8 = (i & 7) << 3;
        const _Float16* rp;
        if (spatial) {
          const int gx = clampi(xw0 + r + j, 0, NXD - 1);
          rp = h16 + (rowc + gx) * DD;
        } else {
          rp = h16 + (nbase + r) * DD;
        }
        *(v8h*)&sINw[r * LD_IN + 64 + c8] = *(const v8h*)(rp + c8);
      }
    } else {
      for (int i = lane; i < 256; i += 32) {
        const int r = i >> 4, c4 = (i & 15) << 2;
        const float* rp;
        if (spatial) {
          const int gx = clampi(xw0 + r + j, 0, NXD - 1);
          rp = h + (rowc + gx) * DD;
        } else {
          rp = h + (nbase + r) * DD;
        }
        v4f f = *(const v4f*)(rp + c4);
        v4h o;
#pragma unroll
        for (int e = 0; e < 4; ++e) o[e] = (_Float16)f[e];
        *(v4h*)&sINw[r * LD_IN + 64 + c4] = o;
      }
    }
    // cols 128..159: scalar features + zero pad (clamped into f16 range)
    for (int i = lane; i < 16 * 32; i += 32) {
      const int r = i >> 5, c = i & 31;
      float f = 0.f;
      if (spatial) {
        const int gx = clampi(xw0 + r + j, 0, NXD - 1);
        const float xi = sX[rg0 + r];
        const float xj = x[b * NXD + gx];
        const float du = sU0[rg0 + r] - u0[b * NXD + gx];
        if (c == 0) f = xi;
        else if (c == 1) f = xj;
        else if (c == 2) f = xj - xi;
        else if (c == 3) f = du;
        else if (c == 4) f = fabsf(du);
      } else {
        if (c == 0) f = ti;
        else if (c == 1) f = tj;
        else if (c == 2) f = reltj;
        else if (c == 3) f = sX[rg0 + r] / fmaxf(ti, 1e-6f);
      }
      f = fminf(fmaxf(f, -F16MAX), F16MAX);
      sINw[r * LD_IN + 128 + c] = (_Float16)f;
    }
    __syncthreads();

    const _Float16* W1 = wf + (spatial ? OFF_SP1 : OFF_TP1);
    const _Float16* W2 = wf + (spatial ? OFF_SP2 : OFF_TP2);
    const _Float16* W3 = wf + (spatial ? OFF_SP3 : OFF_TP3);
    const float* B1 = spatial ? bsp1 : btp1;
    const float* B2 = spatial ? bsp2 : btp2;
    const float* B3 = spatial ? bsp3 : btp3;

    layer_gelu<5, 128>(sINw, LD_IN, W1, 160, B1, sH1w, LD_H1, lane);  // 133->128
    layer_gelu<4, 128>(sH1w, LD_H1, W2, 128, B2, sINw, LD_H2, lane);  // 128->128
    layer_msg<4>(sINw, LD_H2, W3, 128, B3, agg, shkm, lane);          // 128->64 gated
    __syncthreads();
  }

  // update MLP input: [h | agg] (16 x 128 f16, stride LD_H2), agg straight from regs
  for (int i = lane; i < 128; i += 32) {
    const int r = i >> 3, c8 = (i & 7) << 3;
    *(v8h*)&sINw[r * LD_H2 + c8] = *(const v8h*)&sHCw[r * LD_HC + c8];
  }
  {
    const int n0 = lane & 15;
#pragma unroll
    for (int nt = 0; nt < 4; ++nt)
#pragma unroll
      for (int v = 0; v < 8; ++v)
        sINw[(mhi + v) * LD_H2 + 64 + nt * 16 + n0] = (_Float16)agg[nt][v];
  }
  __syncthreads();

  layer_gelu<4, 128>(sINw, LD_H2, wf + OFF_UP1, 128, bup1, sH1w, LD_H1, lane);
  v8f loc[4];
  layer_lin<2>(sHCw, LD_HC, wf + OFF_WL, 64, bwl, loc, lane);      // h_local in regs
  layer_gelu<4, 128>(sH1w, LD_H1, wf + OFF_UP2, 128, bup2, sINw, LD_H2, lane);
  layer_final<4>(sINw, LD_H2, wf + OFF_UP3, 128, bup3, loc, out, pbase, lane);
}

extern "C" void kernel_launch(void* const* d_in, const int* in_sizes, int n_in,
                              void* d_out, int out_size, void* d_ws, size_t ws_size,
                              hipStream_t stream) {
  (void)in_sizes; (void)n_in; (void)out_size;
  // d_in order: h, x, t, u0, shock_indicator, then params pytree leaves with
  // dict keys sorted ('W' < 'sp' < 'tp' < 'up'), each layer (W, b):
  // 5: W.W 6: W.b | 7..12: sp (W1,b1,W2,b2,W3,b3) | 13..18: tp | 19..24: up
  _Float16* wf = (_Float16*)d_ws;

  auto cvt = [&](int idx, unsigned off, int rows, int cols, int ldd) {
    const int total = rows * ldd;
    pinn_wprep_f16<<<(total + 255) / 256, 256, 0, stream>>>(
        (const float*)d_in[idx], wf + off, rows, cols, ldd);
  };
  cvt(7,  OFF_SP1, 128, 133, 160);
  cvt(9,  OFF_SP2, 128, 128, 128);
  cvt(11, OFF_SP3, 64,  128, 128);
  cvt(13, OFF_TP1, 128, 132, 160);
  cvt(15, OFF_TP2, 128, 128, 128);
  cvt(17, OFF_TP3, 64,  128, 128);
  cvt(19, OFF_UP1, 128, 128, 128);
  cvt(21, OFF_UP2, 128, 128, 128);
  cvt(23, OFF_UP3, 64,  128, 128);
  cvt(5,  OFF_WL,  64,  64,  64);

  // optional: pre-convert h to f16 in workspace (kills 10x f32 re-reads + cvts)
  const size_t H_ELEMS = (size_t)NB * NT * NXD * DD;           // 16,777,216
  const size_t NEED = (size_t)2 * (OFF_H16 + H_ELEMS);
  _Float16* h16 = nullptr;
  if (ws_size >= NEED) {
    h16 = wf + OFF_H16;
    const int n4 = (int)(H_ELEMS / 4);
    pinn_hprep_f16<<<(n4 + 255) / 256, 256, 0, stream>>>(
        (const float*)d_in[0], h16, n4);
  }

  const int nblocks = NB * NT * (NXD / 64);  // 4096
  pinn_mp_main<<<nblocks, 128, 0, stream>>>(
      (const float*)d_in[0], (const float*)d_in[1], (const float*)d_in[2],
      (const float*)d_in[3], (const float*)d_in[4],
      wf, h16,
      (const float*)d_in[8],  (const float*)d_in[10], (const float*)d_in[12],
      (const float*)d_in[14], (const float*)d_in[16], (const float*)d_in[18],
      (const float*)d_in[20], (const float*)d_in[22], (const float*)d_in[24],
      (const float*)d_in[6],
      (float*)d_out);
}
